// LSTMmodel_15960098472574
// MI455X (gfx1250) — compile-verified, matrix operand order checked by reading, not run on hardware
//
#include <hip/hip_runtime.h>
#include <hip/hip_bf16.h>
#include <stdint.h>

typedef __attribute__((ext_vector_type(16))) __bf16 v16bf;
typedef __attribute__((ext_vector_type(8)))  float  v8f;
typedef unsigned short u16;

#define T_LEN  256
#define BATCH  16
#define IN_DIM 512
#define H_DIM  1024
#define V_DIM  32000
#define K_DIM  1536           // IN + H
#define LSTM_WGS 16           // persistent grid size
// LDS: 64 weight rows * 1536 * bf16 = 192KB, + 512B wave-reduction scratch
#define LDS_W_BYTES (64 * K_DIM * 2)
#define LDS_BYTES   (LDS_W_BYTES + 512)

// ---- workspace layout (bytes) ----
#define CTR_OFF   0
#define ZSUM_OFF  256
#define ZSQ_OFF   (ZSUM_OFF + T_LEN*BATCH*4)          // 16640
#define INIT_BYTES 33280                               // memset ctr+zsum+zsq
#define HB_OFF    33280                                // h bf16 32KB
#define X_OFF     (HB_OFF  + BATCH*H_DIM*2)           // X bf16 4MB
#define WGO_OFF   (X_OFF   + T_LEN*BATCH*IN_DIM*2)    // Wg[2] bf16 3MB
#define OUTS_OFF  (WGO_OFF + H_DIM*K_DIM*2)           // outs bf16 8MB
#define WOUTB_OFF (OUTS_OFF+ BATCH*T_LEN*H_DIM*2)     // Wout bf16 64MB

__device__ __forceinline__ u16 f2bf(float f) {        // RNE f32 -> bf16
  unsigned u = __float_as_uint(f);
  u += 0x7FFFu + ((u >> 16) & 1u);
  return (u16)(u >> 16);
}

// 16-lane butterfly sum via DPP16 (no LDS, no waits):
// quad_perm(xor1), quad_perm(xor2), ROW_HALF_MIRROR, ROW_MIRROR
__device__ __forceinline__ float dpp_red16(float x) {
  x += __int_as_float(__builtin_amdgcn_update_dpp(0, __float_as_int(x), 0xB1,  0xF, 0xF, true));
  x += __int_as_float(__builtin_amdgcn_update_dpp(0, __float_as_int(x), 0x4E,  0xF, 0xF, true));
  x += __int_as_float(__builtin_amdgcn_update_dpp(0, __float_as_int(x), 0x141, 0xF, 0xF, true));
  x += __int_as_float(__builtin_amdgcn_update_dpp(0, __float_as_int(x), 0x140, 0xF, 0xF, true));
  return x;
}

// ---------------- prep kernels ----------------
__global__ void k_gather_x(const int* __restrict__ tokens,
                           const float* __restrict__ emb, u16* __restrict__ X) {
  int idx = blockIdx.x * 256 + threadIdx.x;            // < T*B*IN
  int k  = idx & (IN_DIM - 1);
  int bt = idx >> 9;                                   // = t*16 + b
  int b  = bt & 15, t = bt >> 4;
  int tok = tokens[b * T_LEN + t];
  float v = (tok == 0) ? 0.0f : emb[(size_t)tok * IN_DIM + k];
  X[idx] = f2bf(v);
}

__global__ void k_cvt_wgo(const float* __restrict__ Wg, u16* __restrict__ Wgo) {
  int idx = blockIdx.x * 256 + threadIdx.x;            // < 1024*1536
  Wgo[idx] = f2bf(Wg[2 * H_DIM * K_DIM + idx]);        // gate 2 = output gate
}

__global__ void k_cvt_wout(const float* __restrict__ W, u16* __restrict__ O) {
  int i = blockIdx.x * 256 + threadIdx.x;              // 4 elems/thread
  float4 v = ((const float4*)W)[i];
  ushort4 u; u.x = f2bf(v.x); u.y = f2bf(v.y); u.z = f2bf(v.z); u.w = f2bf(v.w);
  ((ushort4*)O)[i] = u;
}

__global__ void k_init_h(const float* __restrict__ h0, u16* __restrict__ hb) {
  int idx = blockIdx.x * 256 + threadIdx.x;            // < 16*1024
  hb[idx] = f2bf(h0[idx & (H_DIM - 1)]);
}

// ---------------- grid-wide sync ----------------
// Fast path: CDNA5 cluster split-barrier (if dispatched as one 16-WG cluster).
// Fallback: release/acquire atomic ticket barrier through L2.
__device__ __forceinline__ void grid_sync(unsigned* ctr, unsigned gen, bool clus) {
  __syncthreads();
  __threadfence();
  if (clus) {
    if ((threadIdx.x >> 5) == 0)                       // one wave signals per WG
      asm volatile("s_barrier_signal -3" ::: "memory");
    asm volatile("s_barrier_wait -3" ::: "memory");    // all waves wait
  } else {
    if (threadIdx.x == 0) {
      __hip_atomic_fetch_add(ctr, 1u, __ATOMIC_RELEASE, __HIP_MEMORY_SCOPE_AGENT);
      unsigned tgt = gen * gridDim.x;
      while (__hip_atomic_load(ctr, __ATOMIC_RELAXED, __HIP_MEMORY_SCOPE_AGENT) < tgt)
        __builtin_amdgcn_s_sleep(1);
    }
    __syncthreads();
  }
  (void)__hip_atomic_load(ctr, __ATOMIC_ACQUIRE, __HIP_MEMORY_SCOPE_AGENT);
}

// ---------------- persistent recurrent kernel ----------------
// 16 WGs x 128 threads (4 waves). Wave w of WG g owns N-tile (g*4+w): 16 cols.
// Weights in LDS; A = [X_t | h]; h master state lives in registers (hreg).
__global__ void __launch_bounds__(128) k_lstm(
    const u16* __restrict__ X, const u16* __restrict__ Wgo,
    const float* __restrict__ bg, const float* __restrict__ lns,
    const float* __restrict__ lnb, const float* __restrict__ h0,
    u16* __restrict__ hb, u16* __restrict__ outs,
    float* __restrict__ zsum, float* __restrict__ zsq, unsigned* ctr) {
  extern __shared__ u16 lds_w[];
  float* red_s = (float*)(lds_w + 64 * K_DIM);   // [4 waves][16 m]
  float* red_q = red_s + 64;

  const int tid  = threadIdx.x;
  const int lane = tid & 31;
  const int wave = tid >> 5;
  const int ln15 = lane & 15;
  const int hsel = lane >> 4;            // 0: lanes 0-15, 1: lanes 16-31
  const int o8   = hsel * 8;             // K sub-chunk per ISA A/B layout
  const int n_g  = (blockIdx.x * 4 + wave) * 16 + ln15;  // global column

  // cluster detect: IB_STS2[9:6] != 0 means dispatched inside a cluster
  unsigned cid;
  asm volatile("s_getreg_b32 %0, hwreg(HW_REG_IB_STS2, 6, 4)" : "=s"(cid));
  const bool clus = (cid != 0);

  // stage this WG's 64 weight rows (K-major) into LDS: 192KB
  {
    const uint4* src = (const uint4*)(Wgo + (size_t)blockIdx.x * 64 * K_DIM);
    uint4* dst = (uint4*)lds_w;
    for (int i = tid; i < 64 * K_DIM / 8; i += 128) dst[i] = src[i];
  }
  __syncthreads();

  const float bg_n = bg [2 * H_DIM + n_g];
  const float sc_n = lns[2 * H_DIM + n_g];
  const float lb_n = lnb[2 * H_DIM + n_g];
  const u16* brow  = lds_w + (wave * 16 + ln15) * K_DIM;   // B: column n_g
  const u16* hrow  = hb + (size_t)ln15 * H_DIM;

  // f32 master h: lane owns (m = r + 8*hsel, n_g) forever -> registers
  float hreg[8];
  {
    float h0n = h0[n_g];
    #pragma unroll
    for (int r = 0; r < 8; ++r) hreg[r] = h0n;
  }

  unsigned gen = 0;
  for (int t = 0; t < T_LEN; ++t) {
    v8f acc = {};
    const u16* xrow = X + (size_t)(t * BATCH + ln15) * IN_DIM;
    union Frag { uint4 q[2]; v16bf v; };
    #pragma unroll 4
    for (int kk = 0; kk < IN_DIM / 32; ++kk) {         // x_t part of comb
      const int kb = kk * 32;
      Frag a, b;
      a.q[0] = *(const uint4*)(xrow + kb + o8);
      a.q[1] = *(const uint4*)(xrow + kb + 16 + o8);
      b.q[0] = *(const uint4*)(brow + kb + o8);        // ds_load_b128
      b.q[1] = *(const uint4*)(brow + kb + 16 + o8);
      acc = __builtin_amdgcn_wmma_f32_16x16x32_bf16(
                false, a.v, false, b.v, (short)0, acc, false, false);
    }
    #pragma unroll 4
    for (int kk = 0; kk < H_DIM / 32; ++kk) {          // h part of comb
      const int kb = kk * 32;
      Frag a, b;
      a.q[0] = *(const uint4*)(hrow + kb + o8);
      a.q[1] = *(const uint4*)(hrow + kb + 16 + o8);
      b.q[0] = *(const uint4*)(brow + IN_DIM + kb + o8);
      b.q[1] = *(const uint4*)(brow + IN_DIM + kb + 16 + o8);
      acc = __builtin_amdgcn_wmma_f32_16x16x32_bf16(
                false, a.v, false, b.v, (short)0, acc, false, false);
    }
    // add gate bias (depends only on n) before LN statistics
    #pragma unroll
    for (int r = 0; r < 8; ++r) acc[r] += bg_n;

    // LN partials: DPP16 butterfly over the 16 lanes sharing each M row,
    // then combine the WG's 4 waves in LDS, then ONE atomic per (WG, m).
    #pragma unroll
    for (int r = 0; r < 8; ++r) {
      float s = dpp_red16(acc[r]);
      float q = dpp_red16(acc[r] * acc[r]);
      if (ln15 == 0) {
        int m = r + 8 * hsel;
        red_s[wave * 16 + m] = s;
        red_q[wave * 16 + m] = q;
      }
    }
    __syncthreads();
    if (tid < BATCH) {
      float s = red_s[tid] + red_s[16 + tid] + red_s[32 + tid] + red_s[48 + tid];
      float q = red_q[tid] + red_q[16 + tid] + red_q[32 + tid] + red_q[48 + tid];
      atomicAdd(&zsum[t * BATCH + tid], s);
      atomicAdd(&zsq [t * BATCH + tid], q);
    }
    grid_sync(ctr, ++gen, clus);

    // finalize: LN, sigmoid, h_new = o * h_prev (the only live gate)
    #pragma unroll
    for (int r = 0; r < 8; ++r) {
      int mrow   = r + 8 * hsel;
      float mean = zsum[t * BATCH + mrow] * (1.0f / H_DIM);
      float var  = zsq [t * BATCH + mrow] * (1.0f / H_DIM) - mean * mean;
      float rstd = rsqrtf(var + 1e-5f);
      float zn   = (acc[r] - mean) * rstd * sc_n + lb_n;
      float o    = 1.0f / (1.0f + __expf(-zn));
      float hnew = o * hreg[r];
      hreg[r] = hnew;
      u16 hbf = f2bf(hnew);
      hb[(size_t)mrow * H_DIM + n_g] = hbf;
      outs[((size_t)mrow * T_LEN + t) * H_DIM + n_g] = hbf;   // row = b*T + t
    }
    grid_sync(ctr, ++gen, clus);
  }
}

// ---------------- output projection: [4096,1024] x [32000,1024]^T ----------------
// WG = 256 thr = 8 waves; WG tile 64 rows x 256 cols; wave: 64x32 (4x2 frags).
__global__ void __launch_bounds__(256) k_proj(
    const u16* __restrict__ A, const u16* __restrict__ Bm,
    const float* __restrict__ bout, float* __restrict__ out) {
  const int tid  = threadIdx.x;
  const int lane = tid & 31;
  const int wave = tid >> 5;
  const int ln15 = lane & 15;
  const int o8   = (lane >> 4) * 8;
  const int mbase = blockIdx.y * 64;
  const int nbase = blockIdx.x * 256 + wave * 32;

  v8f acc[4][2] = {};
  for (int kk = 0; kk < H_DIM / 32; ++kk) {
    const int kb = kk * 32;
    union F { uint4 q[2]; v16bf v; };
    F a[4], b[2];
    #pragma unroll
    for (int mt = 0; mt < 4; ++mt) {
      const u16* p = A + (size_t)(mbase + mt * 16 + ln15) * H_DIM + kb + o8;
      a[mt].q[0] = *(const uint4*)p;
      a[mt].q[1] = *(const uint4*)(p + 16);
    }
    #pragma unroll
    for (int nt = 0; nt < 2; ++nt) {
      const u16* p = Bm + (size_t)(nbase + nt * 16 + ln15) * H_DIM + kb + o8;
      b[nt].q[0] = *(const uint4*)p;
      b[nt].q[1] = *(const uint4*)(p + 16);
      __builtin_prefetch(p + 256, 0, 1);               // global_prefetch_b8
    }
    #pragma unroll
    for (int mt = 0; mt < 4; ++mt)
      #pragma unroll
      for (int nt = 0; nt < 2; ++nt)
        acc[mt][nt] = __builtin_amdgcn_wmma_f32_16x16x32_bf16(
            false, a[mt].v, false, b[nt].v, (short)0, acc[mt][nt], false, false);
  }
  #pragma unroll
  for (int mt = 0; mt < 4; ++mt)
    for (int nt = 0; nt < 2; ++nt) {
      int n_g = nbase + nt * 16 + ln15;
      float bo = bout[n_g];
      #pragma unroll
      for (int r = 0; r < 8; ++r) {
        int m_g = mbase + mt * 16 + r + o8;            // C/D layout: M = r + 8*hsel
        out[(size_t)m_g * V_DIM + n_g] = acc[mt][nt][r] + bo;
      }
    }
}

// ---------------- launcher ----------------
extern "C" void kernel_launch(void* const* d_in, const int* in_sizes, int n_in,
                              void* d_out, int out_size, void* d_ws, size_t ws_size,
                              hipStream_t stream) {
  const int*   tokens = (const int*)  d_in[0];
  const float* emb    = (const float*)d_in[1];
  const float* Wg     = (const float*)d_in[2];
  const float* bg     = (const float*)d_in[3];
  const float* lns    = (const float*)d_in[4];
  const float* lnb    = (const float*)d_in[5];
  const float* h0     = (const float*)d_in[6];
  // d_in[7] = c0: dead — c never influences h or the logits
  const float* Wout   = (const float*)d_in[8];
  const float* bout   = (const float*)d_in[9];
  float* out = (float*)d_out;

  char* ws = (char*)d_ws;
  unsigned* ctr = (unsigned*)(ws + CTR_OFF);
  float* zsum   = (float*)(ws + ZSUM_OFF);
  float* zsq    = (float*)(ws + ZSQ_OFF);
  u16*   hb     = (u16*)  (ws + HB_OFF);
  u16*   X      = (u16*)  (ws + X_OFF);
  u16*   Wgo    = (u16*)  (ws + WGO_OFF);
  u16*   outs   = (u16*)  (ws + OUTS_OFF);
  u16*   WoB    = (u16*)  (ws + WOUTB_OFF);

  // re-zero barrier + LN accumulators every call (graph-replay safe)
  hipMemsetAsync(ws, 0, INIT_BYTES, stream);

  hipLaunchKernelGGL(k_gather_x, dim3(T_LEN*BATCH*IN_DIM/256), dim3(256), 0, stream,
                     tokens, emb, X);
  hipLaunchKernelGGL(k_cvt_wgo,  dim3(H_DIM*K_DIM/256), dim3(256), 0, stream, Wg, Wgo);
  hipLaunchKernelGGL(k_cvt_wout, dim3(V_DIM*H_DIM/1024), dim3(256), 0, stream, Wout, WoB);
  hipLaunchKernelGGL(k_init_h,   dim3(BATCH*H_DIM/256), dim3(256), 0, stream, h0, hb);

  hipFuncSetAttribute(reinterpret_cast<const void*>(&k_lstm),
                      hipFuncAttributeMaxDynamicSharedMemorySize, LDS_BYTES);
  hipLaunchKernelGGL(k_lstm, dim3(LSTM_WGS), dim3(128), LDS_BYTES, stream,
                     X, Wgo, bg, lns, lnb, h0, hb, outs, zsum, zsq, ctr);

  hipLaunchKernelGGL(k_proj, dim3(V_DIM/256, (BATCH*T_LEN)/64), dim3(256), 0, stream,
                     outs, WoB, bout, out);
}
